// PositionFeaturizer_49435073577362
// MI455X (gfx1250) — matrix operand order repaired and checked by reading, not compiled, not run on hardware
//
#include <hip/hip_runtime.h>
#include <hip/hip_bf16.h>
#include <stdint.h>

// ---------------------------------------------------------------------------
// PositionFeaturizer for MI455X (gfx1250, wave32, WMMA).
//
// Roofline: ~52 GFLOP of dense GEMM vs ~0.2 GB of HBM traffic (23.3 TB/s).
// f32 VALU would be compute-bound; f16-input WMMA (v_wmma_f32_16x16x32_f16,
// f32 accumulate) brings the GEMMs to the same tens-of-us scale as memory.
// Edge-sparse softmax stays in f32 with L2 atomics (q/k/logits are L2-resident).
// ---------------------------------------------------------------------------

#define N_NODES 16384
#define S_NODES 16384
#define E_EDGES 262144
#define DIM     512
#define H_HEADS 8
#define HD      64
#define HID     1024
#define MLP_IN  536
#define MLP_PAD 544     // pad K to a multiple of 32 for the f16 WMMA

typedef _Float16 f16;
typedef _Float16 v16h __attribute__((ext_vector_type(16)));
typedef float    v8f  __attribute__((ext_vector_type(8)));

// ---- workspace layout (bytes, all regions 16B+ aligned) -------------------
static constexpr size_t OFF_ZATT  = 0;                                          // [N,512] f16
static constexpr size_t OFF_QH    = OFF_ZATT  + (size_t)N_NODES*DIM*2;          // [N,512] f16
static constexpr size_t OFF_KH    = OFF_QH    + (size_t)N_NODES*DIM*2;          // [N,512] f16
static constexpr size_t OFF_MLPIN = OFF_KH    + (size_t)N_NODES*DIM*2;          // [N,544] f16
static constexpr size_t OFF_HID   = OFF_MLPIN + (size_t)N_NODES*MLP_PAD*2;      // [N,1024] f16
static constexpr size_t OFF_WQH   = OFF_HID   + (size_t)N_NODES*HID*2;          // [512,512] f16
static constexpr size_t OFF_WKH   = OFF_WQH   + (size_t)DIM*DIM*2;
static constexpr size_t OFF_WINH  = OFF_WKH   + (size_t)DIM*DIM*2;              // [1024,544] f16
static constexpr size_t OFF_WOUTH = OFF_WINH  + (size_t)HID*MLP_PAD*2;          // [512,1024] f16
static constexpr size_t OFF_LOG   = OFF_WOUTH + (size_t)DIM*HID*2;              // [E,8] f32
static constexpr size_t OFF_RMAX  = OFF_LOG   + (size_t)E_EDGES*H_HEADS*4;      // [N,8] f32
static constexpr size_t OFF_RDEN  = OFF_RMAX  + (size_t)N_NODES*H_HEADS*4;
static constexpr size_t OFF_RSUM  = OFF_RDEN  + (size_t)N_NODES*H_HEADS*4;
static constexpr size_t OFF_FEAT  = OFF_RSUM  + (size_t)N_NODES*H_HEADS*4;      // [N,24] f32

// ---------------------------------------------------------------------------
__device__ __forceinline__ float warp_sum(float v) {
  #pragma unroll
  for (int o = 16; o > 0; o >>= 1) v += __shfl_xor(v, o, 32);
  return v;
}

__device__ __forceinline__ float gelu_tanh(float v) {
  float v3 = v * v * v;
  return 0.5f * v * (1.0f + tanhf(0.7978845608028654f * (v + 0.044715f * v3)));
}

// bit-trick float atomic max (init with -inf); positive -> signed int max,
// negative -> unsigned min. Lowers to global_atomic_max_i32 / _min_u32.
__device__ __forceinline__ void atomic_max_f32(float* addr, float val) {
  if (val >= 0.0f) atomicMax((int*)addr, __float_as_int(val));
  else             atomicMin((unsigned int*)addr, __float_as_uint(val));
}

// ---- WMMA fragment loads (contiguous k = 16h + s order, same for A and B) -------------------
// A 16x32 f16: lanes 0-15 row M=lane, K k0..k0+15;
//              lanes 16-31 same rows, K k0+16..k0+31. Two b128 loads.
__device__ __forceinline__ v16h load_fragA(const f16* __restrict__ base, int ld,
                                           int m, int k0, int lane) {
  int hb = (lane >> 4) << 4;
  const f16* r = base + (size_t)m * ld + k0 + hb;
  union { v16h h; uint4 u[2]; } f;
  f.u[0] = *(const uint4*)(r);
  f.u[1] = *(const uint4*)(r + 8);
  return f.h;
}
// B 32x16 f16: lane holds column n=lane&15; lanes 0-15 K=k0..k0+15,
// lanes 16-31 K=k0+16..k0+31 (16 contiguous K per lane). B[k][n] = W[n][k]
// so varying k is contiguous along the (row-major) weight row.
__device__ __forceinline__ v16h load_fragB(const f16* __restrict__ w, int ld,
                                           int n, int k0, int lane) {
  const f16* r = w + (size_t)n * ld + k0 + ((lane >> 4) << 4);
  union { v16h h; uint4 u[2]; } f;
  f.u[0] = *(const uint4*)(r);
  f.u[1] = *(const uint4*)(r + 8);
  return f.h;
}

struct Acc4 { v8f c0, c1, c2, c3; };

// One wave computes a 16x64 f32 tile of A[M,K] @ W[N,K]^T (A-frag reuse x4).
__device__ __forceinline__ Acc4 wmma_tile_16x64(const f16* __restrict__ A, int lda,
                                                const f16* __restrict__ W, int ldw,
                                                int K, int m, int colbase, int lane) {
  v8f z = {0.f,0.f,0.f,0.f,0.f,0.f,0.f,0.f};
  Acc4 a; a.c0 = z; a.c1 = z; a.c2 = z; a.c3 = z;
  int nl = lane & 15;
  for (int k0 = 0; k0 < K; k0 += 32) {
    v16h af = load_fragA(A, lda, m, k0, lane);
    v16h b0 = load_fragB(W, ldw, colbase +  0 + nl, k0, lane);
    v16h b1 = load_fragB(W, ldw, colbase + 16 + nl, k0, lane);
    v16h b2 = load_fragB(W, ldw, colbase + 32 + nl, k0, lane);
    v16h b3 = load_fragB(W, ldw, colbase + 48 + nl, k0, lane);
    a.c0 = __builtin_amdgcn_wmma_f32_16x16x32_f16(false, af, false, b0, (short)0, a.c0, false, false);
    a.c1 = __builtin_amdgcn_wmma_f32_16x16x32_f16(false, af, false, b1, (short)0, a.c1, false, false);
    a.c2 = __builtin_amdgcn_wmma_f32_16x16x32_f16(false, af, false, b2, (short)0, a.c2, false, false);
    a.c3 = __builtin_amdgcn_wmma_f32_16x16x32_f16(false, af, false, b3, (short)0, a.c3, false, false);
  }
  return a;
}

// ---------------------------------------------------------------------------
// K0: init accumulators (every launch -> deterministic, no cross-call state)
__global__ void k_init(float* rmax, float* rden, float* rsum, float* feat) {
  int i = blockIdx.x * blockDim.x + threadIdx.x;
  if (i < N_NODES * H_HEADS) { rmax[i] = -__builtin_inff(); rden[i] = 0.f; rsum[i] = 0.f; }
  if (i < N_NODES * 24)      { feat[i] = 0.f; }
}

// K1: convert weights to f16 (W_in padded 536 -> 544 with zeros)
__global__ void k_cvt_weights(const float* __restrict__ Wq, const float* __restrict__ Wk,
                              const float* __restrict__ Win, const float* __restrict__ Wout,
                              f16* wq_h, f16* wk_h, f16* win_h, f16* wout_h) {
  int i = blockIdx.x * blockDim.x + threadIdx.x;
  if (i < DIM * DIM) { wq_h[i] = (f16)Wq[i]; wk_h[i] = (f16)Wk[i]; }
  if (i < DIM * HID) { wout_h[i] = (f16)Wout[i]; }
  if (i < HID * MLP_PAD) {
    int r = i / MLP_PAD, c = i - r * MLP_PAD;
    win_h[i] = (c < MLP_IN) ? (f16)Win[r * MLP_IN + c] : (f16)0.f;
  }
}

// K2: dual LayerNorm, wave per row (wave32 shfl reduction), f16 outputs.
__global__ void k_layernorm(const float* __restrict__ x,
                            const float* __restrict__ g_att, const float* __restrict__ b_att,
                            const float* __restrict__ g_mlp, const float* __restrict__ b_mlp,
                            f16* __restrict__ zatt, f16* __restrict__ mlp_in) {
  int lane = threadIdx.x & 31;
  int wave = threadIdx.x >> 5;
  int n = blockIdx.x * 8 + wave;
  if (n >= N_NODES) return;
  const float4* xr = (const float4*)(x + (size_t)n * DIM);
  float4 v[4]; float s = 0.f, sq = 0.f;
  #pragma unroll
  for (int i = 0; i < 4; i++) {
    v[i] = xr[lane + i * 32];
    float* p = (float*)&v[i];
    #pragma unroll
    for (int c = 0; c < 4; c++) { s += p[c]; sq += p[c] * p[c]; }
  }
  s = warp_sum(s); sq = warp_sum(sq);
  float mu = s * (1.0f / DIM);
  float var = sq * (1.0f / DIM) - mu * mu;
  float rs = rsqrtf(var + 1e-5f);
  f16* za = zatt + (size_t)n * DIM;
  f16* zm = mlp_in + (size_t)n * MLP_PAD;
  #pragma unroll
  for (int i = 0; i < 4; i++) {
    float* p = (float*)&v[i];
    int d0 = (lane + i * 32) * 4;
    #pragma unroll
    for (int c = 0; c < 4; c++) {
      float t = (p[c] - mu) * rs;
      za[d0 + c] = (f16)(t * g_att[d0 + c] + b_att[d0 + c]);
      zm[d0 + c] = (f16)(t * g_mlp[d0 + c] + b_mlp[d0 + c]);
    }
  }
}

// K3: Q/K projection GEMM. grid.y 0..7 -> Q columns, 8..15 -> K columns.
__global__ void k_gemm_qk(const f16* __restrict__ zatt,
                          const f16* __restrict__ wqh, const f16* __restrict__ wkh,
                          const float* __restrict__ bq, const float* __restrict__ bk,
                          f16* __restrict__ qh, f16* __restrict__ kh) {
  int lane = threadIdx.x & 31, wave = threadIdx.x >> 5;
  int rowbase = (blockIdx.x * 8 + wave) * 16;
  int cb = blockIdx.y * 64;
  const f16* W; const float* bias; f16* out; int colbase;
  if (cb < DIM) { W = wqh; bias = bq; out = qh; colbase = cb; }
  else          { W = wkh; bias = bk; out = kh; colbase = cb - DIM; }
  int m = rowbase + (lane & 15);
  Acc4 a = wmma_tile_16x64(zatt, DIM, W, DIM, DIM, m, colbase, lane);
  v8f cc[4] = { a.c0, a.c1, a.c2, a.c3 };
  int hi = lane >> 4, nl = lane & 15;
  #pragma unroll
  for (int t = 0; t < 4; t++) {
    int n = colbase + t * 16 + nl;
    float bv = bias[n];
    #pragma unroll
    for (int r = 0; r < 8; r++) {
      int mm = rowbase + hi * 8 + r;
      out[(size_t)mm * DIM + n] = (f16)(cc[t][r] + bv);
    }
  }
}

// K4: per-(edge,head) logits + float atomic segment max
__global__ void k_logits(const f16* __restrict__ qh, const f16* __restrict__ kh,
                         const int* __restrict__ row_index, const int* __restrict__ src_index,
                         const int* __restrict__ org_to_src, const float* __restrict__ att_bias,
                         float* __restrict__ logits, float* rmax) {
  int i = blockIdx.x * blockDim.x + threadIdx.x;
  if (i >= E_EDGES * H_HEADS) return;
  int e = i >> 3, h = i & 7;
  int row  = row_index[e];
  int node = org_to_src[src_index[e]];
  const uint4* qp = (const uint4*)(qh + (size_t)row  * DIM + h * HD);
  const uint4* kp = (const uint4*)(kh + (size_t)node * DIM + h * HD);
  float acc = 0.f;
  #pragma unroll
  for (int j = 0; j < 8; j++) {
    uint4 qa = qp[j], ka = kp[j];
    const f16* qf = (const f16*)&qa; const f16* kf = (const f16*)&ka;
    #pragma unroll
    for (int c = 0; c < 8; c++) acc += (float)qf[c] * (float)kf[c];
  }
  float lg = acc * 0.125f + att_bias[(size_t)h * E_EDGES + e];   // 1/sqrt(64)
  logits[i] = lg;
  atomic_max_f32(&rmax[row * H_HEADS + h], lg);
}

// K5: p = exp(logit - max); segment denominator via atomic add
__global__ void k_expsum(const int* __restrict__ row_index, float* __restrict__ logits,
                         const float* __restrict__ rmax, float* rden) {
  int i = blockIdx.x * blockDim.x + threadIdx.x;
  if (i >= E_EDGES * H_HEADS) return;
  int e = i >> 3, h = i & 7;
  int row = row_index[e];
  float p = __expf(logits[i] - rmax[row * H_HEADS + h]);
  logits[i] = p;
  atomicAdd(&rden[row * H_HEADS + h], p);
}

// K6: att = p/den * 1/dist; scatter att*src_pos and rowsum
__global__ void k_scatter(const int* __restrict__ row_index, const int* __restrict__ src_index,
                          const float* __restrict__ dist, const float* __restrict__ src_pos,
                          const float* __restrict__ logits, const float* __restrict__ rden,
                          float* feat, float* rsum) {
  int i = blockIdx.x * blockDim.x + threadIdx.x;
  if (i >= E_EDGES * H_HEADS) return;
  int e = i >> 3, h = i & 7;
  int row = row_index[e];
  float att = logits[i] / rden[row * H_HEADS + h];
  float d = dist[e];
  att = (d == 0.f) ? 0.f : att / d;
  int s = src_index[e];
  float sx = src_pos[s*3+0], sy = src_pos[s*3+1], sz = src_pos[s*3+2];
  float* fb = feat + (size_t)row * 24 + h * 3;
  atomicAdd(fb + 0, att * sx);
  atomicAdd(fb + 1, att * sy);
  atomicAdd(fb + 2, att * sz);
  atomicAdd(&rsum[row * H_HEADS + h], att);
}

// K7: feat - rowsum*pos -> f16 columns [512,536) of mlp_in; zero pad [536,544)
__global__ void k_featpack(const float* __restrict__ feat, const float* __restrict__ rsum,
                           const float* __restrict__ pos, f16* __restrict__ mlp_in) {
  int i = blockIdx.x * blockDim.x + threadIdx.x;
  if (i >= N_NODES * 32) return;
  int n = i >> 5, j = i & 31;
  f16 v;
  if (j < 24) {
    int h = j / 3, c = j - h * 3;
    v = (f16)(feat[(size_t)n * 24 + j] - rsum[n * H_HEADS + h] * pos[n * 3 + c]);
  } else v = (f16)0.f;
  mlp_in[(size_t)n * MLP_PAD + DIM + j] = v;
}

// K8: hidden = gelu(mlp_in @ W_in^T + b_in)  [K=544 padded]
__global__ void k_gemm_hidden(const f16* __restrict__ mlp_in, const f16* __restrict__ winh,
                              const float* __restrict__ b_in, f16* __restrict__ hid) {
  int lane = threadIdx.x & 31, wave = threadIdx.x >> 5;
  int rowbase = (blockIdx.x * 8 + wave) * 16;
  int colbase = blockIdx.y * 64;
  int m = rowbase + (lane & 15);
  Acc4 a = wmma_tile_16x64(mlp_in, MLP_PAD, winh, MLP_PAD, MLP_PAD, m, colbase, lane);
  v8f cc[4] = { a.c0, a.c1, a.c2, a.c3 };
  int hi = lane >> 4, nl = lane & 15;
  #pragma unroll
  for (int t = 0; t < 4; t++) {
    int n = colbase + t * 16 + nl;
    float bv = b_in[n];
    #pragma unroll
    for (int r = 0; r < 8; r++) {
      int mm = rowbase + hi * 8 + r;
      hid[(size_t)mm * HID + n] = (f16)gelu_tanh(cc[t][r] + bv);
    }
  }
}

// K9: out = x + hidden @ W_out^T + b_out  (f32 store)
__global__ void k_gemm_out(const f16* __restrict__ hid, const f16* __restrict__ wouth,
                           const float* __restrict__ b_out, const float* __restrict__ x,
                           float* __restrict__ out) {
  int lane = threadIdx.x & 31, wave = threadIdx.x >> 5;
  int rowbase = (blockIdx.x * 8 + wave) * 16;
  int colbase = blockIdx.y * 64;
  int m = rowbase + (lane & 15);
  Acc4 a = wmma_tile_16x64(hid, HID, wouth, HID, HID, m, colbase, lane);
  v8f cc[4] = { a.c0, a.c1, a.c2, a.c3 };
  int hi = lane >> 4, nl = lane & 15;
  #pragma unroll
  for (int t = 0; t < 4; t++) {
    int n = colbase + t * 16 + nl;
    float bv = b_out[n];
    #pragma unroll
    for (int r = 0; r < 8; r++) {
      int mm = rowbase + hi * 8 + r;
      size_t idx = (size_t)mm * DIM + n;
      out[idx] = x[idx] + cc[t][r] + bv;
    }
  }
}

// ---------------------------------------------------------------------------
extern "C" void kernel_launch(void* const* d_in, const int* in_sizes, int n_in,
                              void* d_out, int out_size, void* d_ws, size_t ws_size,
                              hipStream_t stream) {
  (void)in_sizes; (void)n_in; (void)out_size; (void)ws_size;
  const float* x        = (const float*)d_in[0];
  const float* Wq       = (const float*)d_in[1];
  const float* bq       = (const float*)d_in[2];
  const float* Wk       = (const float*)d_in[3];
  const float* bk       = (const float*)d_in[4];
  const float* g_att    = (const float*)d_in[5];
  const float* b_att    = (const float*)d_in[6];
  const float* g_mlp    = (const float*)d_in[7];
  const float* b_mlp    = (const float*)d_in[8];
  const float* W_in     = (const float*)d_in[9];
  const float* b_in     = (const float*)d_in[10];
  const float* W_out    = (const float*)d_in[11];
  const float* b_out    = (const float*)d_in[12];
  const float* att_bias = (const float*)d_in[13];
  const float* dist     = (const float*)d_in[14];
  const float* pos      = (const float*)d_in[15];
  const float* src_pos  = (const float*)d_in[16];
  const int*   row_idx  = (const int*)d_in[17];
  const int*   src_idx  = (const int*)d_in[18];
  const int*   o2s      = (const int*)d_in[19];
  float* out = (float*)d_out;

  char* ws = (char*)d_ws;
  f16*   zatt   = (f16*)(ws + OFF_ZATT);
  f16*   qh     = (f16*)(ws + OFF_QH);
  f16*   kh     = (f16*)(ws + OFF_KH);
  f16*   mlpin  = (f16*)(ws + OFF_MLPIN);
  f16*   hid    = (f16*)(ws + OFF_HID);
  f16*   wq_h   = (f16*)(ws + OFF_WQH);
  f16*   wk_h   = (f16*)(ws + OFF_WKH);
  f16*   win_h  = (f16*)(ws + OFF_WINH);
  f16*   wout_h = (f16*)(ws + OFF_WOUTH);
  float* logits = (float*)(ws + OFF_LOG);
  float* rmax   = (float*)(ws + OFF_RMAX);
  float* rden   = (float*)(ws + OFF_RDEN);
  float* rsum   = (float*)(ws + OFF_RSUM);
  float* feat   = (float*)(ws + OFF_FEAT);

  const int EH = E_EDGES * H_HEADS;

  k_init<<<(N_NODES * 24 + 255) / 256, 256, 0, stream>>>(rmax, rden, rsum, feat);
  k_cvt_weights<<<(HID * MLP_PAD + 255) / 256, 256, 0, stream>>>(
      Wq, Wk, W_in, W_out, wq_h, wk_h, win_h, wout_h);
  k_layernorm<<<N_NODES / 8, 256, 0, stream>>>(
      x, g_att, b_att, g_mlp, b_mlp, zatt, mlpin);
  k_gemm_qk<<<dim3(N_NODES / 128, 16), 256, 0, stream>>>(
      zatt, wq_h, wk_h, bq, bk, qh, kh);
  k_logits<<<(EH + 255) / 256, 256, 0, stream>>>(
      qh, kh, row_idx, src_idx, o2s, att_bias, logits, rmax);
  k_expsum<<<(EH + 255) / 256, 256, 0, stream>>>(row_idx, logits, rmax, rden);
  k_scatter<<<(EH + 255) / 256, 256, 0, stream>>>(
      row_idx, src_idx, dist, src_pos, logits, rden, feat, rsum);
  k_featpack<<<(N_NODES * 32 + 255) / 256, 256, 0, stream>>>(feat, rsum, pos, mlpin);
  k_gemm_hidden<<<dim3(N_NODES / 128, 16), 256, 0, stream>>>(mlpin, win_h, b_in, hid);
  k_gemm_out<<<dim3(N_NODES / 128, 8), 256, 0, stream>>>(hid, wout_h, b_out, x, out);
}
